// NodePointerEncoder_4913442586878
// MI455X (gfx1250) — compile-verified, matrix operand order checked by reading, not run on hardware
//
#include <hip/hip_runtime.h>
#include <hip/hip_bf16.h>

#define HIDDEN 128
#define LDSA_STRIDE 132   // 128 + 4 pad: rows offset by 4 banks -> conflict-free b64 LDS reads

typedef __attribute__((ext_vector_type(2)))  float  v2f;
typedef __attribute__((ext_vector_type(8)))  float  v8f;
typedef __attribute__((ext_vector_type(16))) __bf16 v16bf;

// Force the hardware f32 atomic-add (no CAS loop). unsafeAtomicAdd is HIP's
// AMD intrinsic that carries the unsafe-fp-atomics attribute.
__device__ __forceinline__ void hw_atomic_add_f32(float* addr, float v) {
#if defined(__HIP_DEVICE_COMPILE__)
    unsafeAtomicAdd(addr, v);
#else
    atomicAdd(addr, v);
#endif
}

// ---------------- Phase 0: zero the aggregation workspace ----------------
__global__ __launch_bounds__(256) void gnn_zero_f4(float4* __restrict__ p, int n4) {
    int i = blockIdx.x * blockDim.x + threadIdx.x;
    if (i < n4) p[i] = make_float4(0.f, 0.f, 0.f, 0.f);
}

// ---------------- Phase A: per-edge gather * prob -> atomic scatter ------
// One wave32 per edge: lane loads float4 of hidden[tgt] (512B coalesced per
// wave, L2-resident), scales by prob[e], atomic-adds into agg[src] with
// no-return global_atomic_add_f32 (STOREcnt path, L2 atomic units).
__global__ __launch_bounds__(256) void gnn_edge_scatter(
    const float* __restrict__ prob,
    const float* __restrict__ hidden,
    const int*   __restrict__ eidx,   // [2, E]: row0 = src (segment), row1 = tgt (gather)
    float*       __restrict__ agg,
    int nE) {
    int lane = threadIdx.x & 31;
    int e    = (blockIdx.x * blockDim.x + threadIdx.x) >> 5;
    if (e >= nE) return;
    int   src = eidx[e];
    int   tgt = eidx[nE + e];
    float p   = prob[e];
    const float4* h4 = (const float4*)(hidden + (size_t)tgt * HIDDEN);
    float4 v = h4[lane];
    float* dst = agg + (size_t)src * HIDDEN + lane * 4;
    hw_atomic_add_f32(dst + 0, v.x * p);
    hw_atomic_add_f32(dst + 1, v.y * p);
    hw_atomic_add_f32(dst + 2, v.z * p);
    hw_atomic_add_f32(dst + 3, v.w * p);
}

__device__ __forceinline__ __bf16 f32_to_bf16(float f) { return (__bf16)f; }

// ---------------- Phase B: out = agg @ W^T + b via WMMA ------------------
// Block = 256 threads = 8 waves. Block computes 16 rows x 128 cols; wave w
// owns the 16-col tile n0 = 16*w. A tile staged in padded LDS; W (64 KB,
// read-only) stays in WGP$/L2.
__global__ __launch_bounds__(256) void gnn_linear_wmma(
    const float* __restrict__ A,     // [M, 128] aggregated
    const float* __restrict__ W,     // [128, 128] row-major [out, in]
    const float* __restrict__ bias,  // [128]
    float*       __restrict__ out,   // [M, 128]
    int M) {
    __shared__ float As[16 * LDSA_STRIDE];

    const int m0 = blockIdx.x * 16;
    const int t  = threadIdx.x;

    // Cooperative coalesced load of the 16x128 f32 A tile into LDS.
    for (int i = t; i < 16 * (HIDDEN / 4); i += 256) {
        int r  = i >> 5;          // 32 float4 per row
        int c4 = i & 31;
        float4 v = ((const float4*)(A + (size_t)(m0 + r) * HIDDEN))[c4];
        float* d = &As[r * LDSA_STRIDE + c4 * 4];
        d[0] = v.x; d[1] = v.y; d[2] = v.z; d[3] = v.w;
    }
    __syncthreads();

    const int wave = t >> 5;
    const int lane = t & 31;
    const int n0   = wave * 16;
    const int lm   = lane & 15;   // M index (A) / N index (B,C,D)
    const int hi   = lane >> 4;   // half-wave select

    const float* Arow = &As[lm * LDSA_STRIDE];               // A row m = m0+lm
    const float* Wrow = W + (size_t)(n0 + lm) * HIDDEN;      // B[k][n] = W[n][k]

    v8f acc = {};

#if __has_builtin(__builtin_amdgcn_wmma_f32_16x16x4_f32)
    // fp32 WMMA: A 16x4 layout -> lane holds M=lm, K = hi*2 + {0,1}.
    const int kb = hi * 2;
    #pragma unroll
    for (int k = 0; k < HIDDEN; k += 4) {
        v2f a, b;
        a.x = Arow[k + kb]; a.y = Arow[k + kb + 1];
        b.x = Wrow[k + kb]; b.y = Wrow[k + kb + 1];
        acc = __builtin_amdgcn_wmma_f32_16x16x4_f32(
            /*neg_a=*/false, a, /*neg_b=*/false, b,
            /*c_mod=*/(short)0, acc, /*reuse_a=*/false, /*reuse_b=*/false);
    }
#else
    // Fallback: bf16 16x16x32 WMMA (codegen-confirmed), fp32 accumulate.
    // 16-bit A 16x32 layout: VGPR j<4 -> K = hi*8 + 2j; j>=4 -> K = 16 + hi*8 + 2(j-4).
    const int koff = hi * 8;
    #pragma unroll
    for (int k = 0; k < HIDDEN; k += 32) {
        v16bf a, b;
        #pragma unroll
        for (int j = 0; j < 8; ++j) {
            int kk = k + ((j < 4) ? (koff + 2 * j) : (16 + koff + 2 * (j - 4)));
            a[2 * j]     = f32_to_bf16(Arow[kk]);
            a[2 * j + 1] = f32_to_bf16(Arow[kk + 1]);
            b[2 * j]     = f32_to_bf16(Wrow[kk]);
            b[2 * j + 1] = f32_to_bf16(Wrow[kk + 1]);
        }
        acc = __builtin_amdgcn_wmma_f32_16x16x32_bf16(
            false, a, false, b, (short)0, acc, false, false);
    }
#endif

    // C/D layout: VGPR r -> M = r + 8*hi, N = lm. Bias is per-N.
    const float bv = bias[n0 + lm];
    #pragma unroll
    for (int r = 0; r < 8; ++r) {
        int m = m0 + r + 8 * hi;
        if (m < M) out[(size_t)m * HIDDEN + n0 + lm] = acc[r] + bv;
    }
}

extern "C" void kernel_launch(void* const* d_in, const int* in_sizes, int n_in,
                              void* d_out, int out_size, void* d_ws, size_t ws_size,
                              hipStream_t stream) {
    const float* prob   = (const float*)d_in[0];   // [E]
    const float* hidden = (const float*)d_in[1];   // [N, 128]
    const int*   eidx   = (const int*)  d_in[2];   // [2, E]
    const float* lin_w  = (const float*)d_in[3];   // [128, 128]
    const float* lin_b  = (const float*)d_in[4];   // [128]
    float* out = (float*)d_out;
    float* agg = (float*)d_ws;                     // [N, 128] scratch accumulator

    const int E = in_sizes[0];
    const int N = in_sizes[1] / HIDDEN;

    // Phase 0: zero accumulator (N*128 floats).
    {
        int n4 = (N * HIDDEN) / 4;
        int blocks = (n4 + 255) / 256;
        gnn_zero_f4<<<blocks, 256, 0, stream>>>((float4*)agg, n4);
    }
    // Phase A: one wave per edge -> E waves -> E/8 blocks of 256.
    {
        int blocks = (E + 7) / 8;
        gnn_edge_scatter<<<blocks, 256, 0, stream>>>(prob, hidden, eidx, agg, E);
    }
    // Phase B: 16 output rows per block.
    {
        int blocks = (N + 15) / 16;
        gnn_linear_wmma<<<blocks, 256, 0, stream>>>(agg, lin_w, lin_b, out, N);
    }
}
